// ActorModel_69664369541581
// MI455X (gfx1250) — matrix operand order repaired and checked by reading, not compiled, not run on hardware
//
#include <hip/hip_runtime.h>
#include <math.h>

// ---------------- problem constants ----------------
constexpr int cB = 64, cS = 256, cH = 768, cL = 12, cNH = 12;
constexpr int cA = 64, cNL = 2, cHD = 64, cH4 = 3072;

typedef __bf16 bf16;
typedef __attribute__((ext_vector_type(16))) __bf16 v16bf;
typedef __attribute__((ext_vector_type(8)))  __bf16 v8bf;
typedef __attribute__((ext_vector_type(8)))  float  v8f;

union Frag { v16bf v; v8bf h[2]; };

// ---------------- GEMM (bf16 WMMA, fp32 accum) ----------------
// C[M,N] = A[M,K] @ Bt[N,K]^T   (Bt stored N-major, K contiguous)
// block tile 128x64, BK=32, 256 threads = 8 waves,
// wave -> 2x2 quad of 16x16 tiles (4 WMMA / K-step, A/B frag reuse)
constexpr int BM = 128, BN = 64, BK = 32;
constexpr int LDP = 40;                        // padded LDS row (halfs) = 80B
constexpr unsigned ABUF = (unsigned)BM * LDP * 2;  // 10240B per A buffer
constexpr unsigned BBUF = (unsigned)BN * LDP * 2;  // 5120B per B buffer

enum GemmMode { MODE_QK = 0, MODE_VT = 1, MODE_SCORES = 2,
                MODE_CTX = 3, MODE_RESF32 = 4, MODE_GELU = 5 };

// CDNA5 async Global->LDS (ASYNCcnt), GVS mode: SGPR64 base + VGPR32 offset
__device__ __forceinline__ void async_ld_b128(unsigned lds_off, unsigned goff,
                                              const void* base)
{
    asm volatile("global_load_async_to_lds_b128 %0, %1, %2"
                 :: "v"(lds_off), "v"(goff), "s"(base) : "memory");
}
__device__ __forceinline__ void wait_async0()
{
    asm volatile("s_wait_asynccnt 0" ::: "memory");
}

template <int MODE>
__launch_bounds__(256)
__global__ void gemm_kernel(const bf16* __restrict__ A, int lda, long long sA,
                            const bf16* __restrict__ Bt, int ldb, long long sB,
                            const float* __restrict__ bias,
                            const float* __restrict__ Res,
                            const int*  __restrict__ mask,
                            float* __restrict__ Cf, bf16* __restrict__ Cb,
                            int K, int ldc, float scale)
{
    __shared__ bf16 As[2][BM][LDP];
    __shared__ bf16 Bs[2][BN][LDP];

    const int z = blockIdx.z;
    A  += (long long)z * sA;
    Bt += (long long)z * sB;

    const int m0 = blockIdx.y * BM;
    const int n0 = blockIdx.x * BN;

    const int tid  = threadIdx.x;
    const int lane = tid & 31;
    const int wid  = tid >> 5;
    const int l16  = lane & 15;
    const int hl   = lane >> 4;              // 0 | 1 (lane half)

    const int mbase = (wid >> 1) * 32;       // 4 M groups of 32
    const int nbase = (wid & 1) * 32;        // 2 N groups of 32

    // LDS byte offsets of the staging buffers (low 32 bits of generic addr)
    const unsigned ldsA = (unsigned)(uintptr_t)&As[0][0][0];
    const unsigned ldsB = (unsigned)(uintptr_t)&Bs[0][0][0];

    // cooperative staging: A tile 128x32 halfs = 512 x 16B (2/thread),
    //                      B tile  64x32 halfs = 256 x 16B (1/thread)
    const unsigned sRow = (unsigned)tid >> 2;      // 0..63
    const unsigned sSeg = (unsigned)(tid & 3);     // 16B segment in row
    const unsigned aG0 = (unsigned)((((size_t)m0 + sRow)      * lda + sSeg * 8) * 2);
    const unsigned aG1 = (unsigned)((((size_t)m0 + sRow + 64) * lda + sSeg * 8) * 2);
    const unsigned bG  = (unsigned)((((size_t)n0 + sRow)      * ldb + sSeg * 8) * 2);
    const unsigned aL0 = ldsA + sRow * 80 + sSeg * 16;
    const unsigned aL1 = ldsA + (sRow + 64) * 80 + sSeg * 16;
    const unsigned bL  = ldsB + sRow * 80 + sSeg * 16;

    // prologue: fill buffer 0
    async_ld_b128(aL0, aG0, A);
    async_ld_b128(aL1, aG1, A);
    async_ld_b128(bL,  bG,  Bt);
    wait_async0();
    __syncthreads();

    v8f acc[4] = {};            // [mi*2 + ni]
    const int nk = K / BK;

    for (int kt = 0; kt < nk; ++kt) {
        const int cur = kt & 1;
        if (kt + 1 < nk) {      // overlap: issue async fills of the other buffer
            const unsigned ko = (unsigned)(kt + 1) * (BK * 2);
            const unsigned oa = (unsigned)((kt + 1) & 1) * ABUF;
            const unsigned ob = (unsigned)((kt + 1) & 1) * BBUF;
            async_ld_b128(aL0 + oa, aG0 + ko, A);
            async_ld_b128(aL1 + oa, aG1 + ko, A);
            async_ld_b128(bL  + ob, bG  + ko, Bt);
            if (kt + 2 < nk) {
                __builtin_prefetch((const char*)A  + aG0 + ko + BK * 2, 0, 1);
                __builtin_prefetch((const char*)Bt + bG  + ko + BK * 2, 0, 1);
            }
        }

        // B fragments: lane n=l16, K = hl*16 .. hl*16+15 (contiguous k order, same as A)
        Frag bf0, bf1;
        bf0.h[0] = *(const v8bf*)&Bs[cur][nbase + l16][hl * 16];
        bf0.h[1] = *(const v8bf*)&Bs[cur][nbase + l16][hl * 16 + 8];
        bf1.h[0] = *(const v8bf*)&Bs[cur][nbase + 16 + l16][hl * 16];
        bf1.h[1] = *(const v8bf*)&Bs[cur][nbase + 16 + l16][hl * 16 + 8];
        // A fragments: lanes0-15 K=0..15, lanes16-31 K=16..31
        Frag af0, af1;
        af0.h[0] = *(const v8bf*)&As[cur][mbase + l16][hl * 16];
        af0.h[1] = *(const v8bf*)&As[cur][mbase + l16][hl * 16 + 8];
        af1.h[0] = *(const v8bf*)&As[cur][mbase + 16 + l16][hl * 16];
        af1.h[1] = *(const v8bf*)&As[cur][mbase + 16 + l16][hl * 16 + 8];

        acc[0] = __builtin_amdgcn_wmma_f32_16x16x32_bf16(false, af0.v, false, bf0.v,
                                                         (short)0, acc[0], false, false);
        acc[1] = __builtin_amdgcn_wmma_f32_16x16x32_bf16(false, af0.v, false, bf1.v,
                                                         (short)0, acc[1], false, false);
        acc[2] = __builtin_amdgcn_wmma_f32_16x16x32_bf16(false, af1.v, false, bf0.v,
                                                         (short)0, acc[2], false, false);
        acc[3] = __builtin_amdgcn_wmma_f32_16x16x32_bf16(false, af1.v, false, bf1.v,
                                                         (short)0, acc[3], false, false);

        if (kt + 1 < nk) {
            wait_async0();      // my async fills landed in LDS
            __syncthreads();    // everyone's landed; prev-buffer reads done
        }
    }

    // ---------------- epilogue ----------------
#pragma unroll
    for (int mi = 0; mi < 2; ++mi) {
#pragma unroll
        for (int ni = 0; ni < 2; ++ni) {
            const v8f a = acc[mi * 2 + ni];
            const int n_g = n0 + nbase + ni * 16 + l16;
#pragma unroll
            for (int r = 0; r < 8; ++r) {
                const int m_g = m0 + mbase + mi * 16 + hl * 8 + r; // C layout 7.12.2
                float v = a[r];
                if constexpr (MODE == MODE_QK) {
                    v += bias[n_g];
                    const int b = m_g / cS, s = m_g % cS;
                    const int hh = n_g >> 6, hd = n_g & 63;
                    Cb[(((size_t)(b * cNH + hh)) * cS + s) * cHD + hd] = (bf16)v;
                } else if constexpr (MODE == MODE_VT) {
                    v += bias[n_g];
                    const int b = m_g / cS, s = m_g % cS;
                    const int hh = n_g >> 6, hd = n_g & 63;
                    Cb[(((size_t)(b * cNH + hh)) * cHD + hd) * cS + s] = (bf16)v;
                } else if constexpr (MODE == MODE_SCORES) {
                    const int b = z / cNH;
                    const float mb = (1.0f - (float)mask[b * cS + n_g]) * -10000.0f;
                    Cf[(size_t)z * cS * cS + (size_t)m_g * cS + n_g] = v * scale + mb;
                } else if constexpr (MODE == MODE_CTX) {
                    const int b = z / cNH, hh = z % cNH;
                    Cb[((size_t)(b * cS + m_g)) * cH + hh * cHD + n_g] = (bf16)v;
                } else if constexpr (MODE == MODE_RESF32) {
                    const size_t idx = (size_t)m_g * ldc + n_g;
                    Cf[idx] = v + bias[n_g] + Res[idx];
                } else { // MODE_GELU
                    const float x = v + bias[n_g];
                    Cb[(size_t)m_g * ldc + n_g] =
                        (bf16)(0.5f * x * (1.0f + erff(x * 0.70710678118654752f)));
                }
            }
        }
    }
}

// ---------------- weight convert + transpose: fp32[R,C] -> bf16[C,R] ----------------
__launch_bounds__(256)
__global__ void tconv_kernel(const float* __restrict__ src, bf16* __restrict__ dst,
                             int R, int C)
{
    __shared__ float tile[32][33];
    const size_t zoff = (size_t)blockIdx.z * R * C;
    src += zoff; dst += zoff;
    const int cb = blockIdx.x * 32, rb = blockIdx.y * 32;
    const int tx = threadIdx.x, ty = threadIdx.y; // (32,8)
#pragma unroll
    for (int i = 0; i < 32; i += 8)
        tile[ty + i][tx] = src[(size_t)(rb + ty + i) * C + cb + tx];
    __syncthreads();
#pragma unroll
    for (int i = 0; i < 32; i += 8)
        dst[(size_t)(cb + ty + i) * R + rb + tx] = (bf16)tile[tx][ty + i];
}

// ---------------- embeddings ----------------
__launch_bounds__(256)
__global__ void embed_kernel(const int* __restrict__ ids, const int* __restrict__ tt,
                             const float* __restrict__ we, const float* __restrict__ pe,
                             const float* __restrict__ te, float* __restrict__ out)
{
    const size_t tok = blockIdx.x;
    const int s = (int)(tok % cS);
    const int id = ids[tok], t = tt[tok];
    const float* w  = we + (size_t)id * cH;
    const float* p  = pe + (size_t)s  * cH;
    const float* ty = te + (size_t)t  * cH;
    float* o = out + tok * cH;
    for (int i = threadIdx.x; i < cH; i += 256) o[i] = w[i] + p[i] + ty[i];
}

// ---------------- layernorm (fp32 in -> fp32 + bf16 out) ----------------
__launch_bounds__(256)
__global__ void ln_kernel(const float* __restrict__ x, const float* __restrict__ sc,
                          const float* __restrict__ bs, float* __restrict__ y,
                          bf16* __restrict__ yb)
{
    const size_t row = blockIdx.x;
    const float* xr = x + row * cH;
    const int tid = threadIdx.x;
    const float v0 = xr[tid], v1 = xr[tid + 256], v2 = xr[tid + 512];
    __shared__ float r1[256], r2[256];
    r1[tid] = v0 + v1 + v2;
    r2[tid] = v0 * v0 + v1 * v1 + v2 * v2;
    __syncthreads();
    for (int st = 128; st > 0; st >>= 1) {
        if (tid < st) { r1[tid] += r1[tid + st]; r2[tid] += r2[tid + st]; }
        __syncthreads();
    }
    const float mean = r1[0] * (1.0f / cH);
    const float var  = r2[0] * (1.0f / cH) - mean * mean;
    const float inv  = rsqrtf(var + 1e-12f);
    float* yr = y + row * cH;
    bf16* ybr = yb + row * cH;
    const float o0 = (v0 - mean) * inv * sc[tid]       + bs[tid];
    const float o1 = (v1 - mean) * inv * sc[tid + 256] + bs[tid + 256];
    const float o2 = (v2 - mean) * inv * sc[tid + 512] + bs[tid + 512];
    yr[tid] = o0;       ybr[tid] = (bf16)o0;
    yr[tid + 256] = o1; ybr[tid + 256] = (bf16)o1;
    yr[tid + 512] = o2; ybr[tid + 512] = (bf16)o2;
}

// ---------------- softmax over rows of 256 (fp32 in -> bf16 out) ----------------
__launch_bounds__(256)
__global__ void softmax_kernel(const float* __restrict__ sc, bf16* __restrict__ attn)
{
    const size_t row = blockIdx.x;
    const float* x = sc + row * cS;
    const int tid = threadIdx.x;
    const float v = x[tid];
    __shared__ float red[256];
    red[tid] = v; __syncthreads();
    for (int st = 128; st > 0; st >>= 1) {
        if (tid < st) red[tid] = fmaxf(red[tid], red[tid + st]);
        __syncthreads();
    }
    const float mx = red[0]; __syncthreads();
    const float e = __expf(v - mx);
    red[tid] = e; __syncthreads();
    for (int st = 128; st > 0; st >>= 1) {
        if (tid < st) red[tid] += red[tid + st];
        __syncthreads();
    }
    attn[row * cS + tid] = (bf16)(e / red[0]);
}

// ---------------- per-annotator head ----------------
__launch_bounds__(64)
__global__ void head_kernel(const float* __restrict__ h, const int* __restrict__ aidx,
                            const float* __restrict__ hw, const float* __restrict__ hb,
                            float* __restrict__ out)
{
    const int b = blockIdx.x;
    const int a = aidx[b];
    const float* cls = h + (size_t)b * cS * cH; // row s=0
    const float* W = hw + (size_t)a * cH * cNL;
    __shared__ float red[64];
    for (int l = 0; l < cNL; ++l) {
        float p = 0.f;
        for (int i = threadIdx.x; i < cH; i += 64) p += cls[i] * W[(size_t)i * cNL + l];
        red[threadIdx.x] = p; __syncthreads();
        for (int st = 32; st > 0; st >>= 1) {
            if (threadIdx.x < st) red[threadIdx.x] += red[threadIdx.x + st];
            __syncthreads();
        }
        if (threadIdx.x == 0) out[b * cNL + l] = red[0] + hb[a * cNL + l];
        __syncthreads();
    }
}

// ---------------- host side ----------------
extern "C" void kernel_launch(void* const* d_in, const int* in_sizes, int n_in,
                              void* d_out, int out_size, void* d_ws, size_t ws_size,
                              hipStream_t stream)
{
    const int*   input_ids = (const int*)d_in[0];
    const int*   attn_mask = (const int*)d_in[1];
    const int*   tok_type  = (const int*)d_in[2];
    const int*   annot     = (const int*)d_in[3];
    const float* word_emb  = (const float*)d_in[4];
    const float* pos_emb   = (const float*)d_in[5];
    const float* type_emb  = (const float*)d_in[6];
    const float* emb_ln_s  = (const float*)d_in[7];
    const float* emb_ln_b  = (const float*)d_in[8];
    const float* Wq = (const float*)d_in[9];   const float* bq = (const float*)d_in[10];
    const float* Wk = (const float*)d_in[11];  const float* bk = (const float*)d_in[12];
    const float* Wv = (const float*)d_in[13];  const float* bv = (const float*)d_in[14];
    const float* Wo = (const float*)d_in[15];  const float* bo = (const float*)d_in[16];
    const float* ln1s = (const float*)d_in[17]; const float* ln1b = (const float*)d_in[18];
    const float* W1 = (const float*)d_in[19];  const float* b1 = (const float*)d_in[20];
    const float* W2 = (const float*)d_in[21];  const float* b2 = (const float*)d_in[22];
    const float* ln2s = (const float*)d_in[23]; const float* ln2b = (const float*)d_in[24];
    const float* head_W = (const float*)d_in[25];
    const float* head_b = (const float*)d_in[26];
    float* out = (float*)d_out;

    size_t off = 0;
    auto carve = [&](size_t bytes) -> void* {
        off = (off + 255) & ~(size_t)255;
        void* p = (char*)d_ws + off;
        off += bytes;
        return p;
    };
    const size_t HH = (size_t)cH * cH, H4H = (size_t)cH * cH4;
    const size_t TOK = (size_t)cB * cS;

    bf16* wtq = (bf16*)carve(cL * HH * 2);
    bf16* wtk = (bf16*)carve(cL * HH * 2);
    bf16* wtv = (bf16*)carve(cL * HH * 2);
    bf16* wto = (bf16*)carve(cL * HH * 2);
    bf16* wt1 = (bf16*)carve(cL * H4H * 2);
    bf16* wt2 = (bf16*)carve(cL * H4H * 2);
    float* tmp  = (float*)carve(TOK * cH * 4);
    float* hf   = (float*)carve(TOK * cH * 4);
    bf16*  hb_  = (bf16*)carve(TOK * cH * 2);
    bf16*  qb   = (bf16*)carve(TOK * cH * 2);
    bf16*  kb   = (bf16*)carve(TOK * cH * 2);
    bf16*  vtb  = (bf16*)carve(TOK * cH * 2);
    float* scf  = (float*)carve((size_t)cB * cNH * cS * cS * 4);
    bf16*  atb  = (bf16*)carve((size_t)cB * cNH * cS * cS * 2);
    bf16*  ctxb = (bf16*)carve(TOK * cH * 2);
    bf16*  ffnb = (bf16*)carve(TOK * cH4 * 2);

    const dim3 tblk(32, 8, 1);
    tconv_kernel<<<dim3(cH / 32, cH / 32, cL), tblk, 0, stream>>>(Wq, wtq, cH, cH);
    tconv_kernel<<<dim3(cH / 32, cH / 32, cL), tblk, 0, stream>>>(Wk, wtk, cH, cH);
    tconv_kernel<<<dim3(cH / 32, cH / 32, cL), tblk, 0, stream>>>(Wv, wtv, cH, cH);
    tconv_kernel<<<dim3(cH / 32, cH / 32, cL), tblk, 0, stream>>>(Wo, wto, cH, cH);
    tconv_kernel<<<dim3(cH4 / 32, cH / 32, cL), tblk, 0, stream>>>(W1, wt1, cH, cH4);
    tconv_kernel<<<dim3(cH / 32, cH4 / 32, cL), tblk, 0, stream>>>(W2, wt2, cH4, cH);

    embed_kernel<<<dim3((unsigned)TOK), 256, 0, stream>>>(input_ids, tok_type, word_emb,
                                                          pos_emb, type_emb, tmp);
    ln_kernel<<<dim3((unsigned)TOK), 256, 0, stream>>>(tmp, emb_ln_s, emb_ln_b, hf, hb_);

    const dim3 gW(cH / BN, (unsigned)(TOK / BM), 1);       // weight GEMMs N=768
    const dim3 gF1(cH4 / BN, (unsigned)(TOK / BM), 1);     // FFN1 N=3072
    const dim3 gSc(cS / BN, cS / BM, cB * cNH);            // scores (4,2,768)
    const dim3 gCtx(cHD / BN, cS / BM, cB * cNH);          // ctx (1,2,768)

    for (int l = 0; l < cL; ++l) {
        const bf16* wq_ = wtq + (size_t)l * HH;
        const bf16* wk_ = wtk + (size_t)l * HH;
        const bf16* wv_ = wtv + (size_t)l * HH;
        const bf16* wo_ = wto + (size_t)l * HH;
        const bf16* w1_ = wt1 + (size_t)l * H4H;
        const bf16* w2_ = wt2 + (size_t)l * H4H;

        gemm_kernel<MODE_QK><<<gW, 256, 0, stream>>>(hb_, cH, 0, wq_, cH, 0,
            bq + (size_t)l * cH, nullptr, nullptr, nullptr, qb, cH, cH, 1.0f);
        gemm_kernel<MODE_QK><<<gW, 256, 0, stream>>>(hb_, cH, 0, wk_, cH, 0,
            bk + (size_t)l * cH, nullptr, nullptr, nullptr, kb, cH, cH, 1.0f);
        gemm_kernel<MODE_VT><<<gW, 256, 0, stream>>>(hb_, cH, 0, wv_, cH, 0,
            bv + (size_t)l * cH, nullptr, nullptr, nullptr, vtb, cH, cH, 1.0f);

        gemm_kernel<MODE_SCORES><<<gSc, 256, 0, stream>>>(
            qb, cHD, (long long)cS * cHD, kb, cHD, (long long)cS * cHD,
            nullptr, nullptr, attn_mask, scf, nullptr, cHD, cS, 0.125f);

        softmax_kernel<<<dim3((unsigned)((size_t)cB * cNH * cS)), 256, 0, stream>>>(scf, atb);

        gemm_kernel<MODE_CTX><<<gCtx, 256, 0, stream>>>(
            atb, cS, (long long)cS * cS, vtb, cS, (long long)cHD * cS,
            nullptr, nullptr, nullptr, nullptr, ctxb, cS, cHD, 1.0f);

        gemm_kernel<MODE_RESF32><<<gW, 256, 0, stream>>>(ctxb, cH, 0, wo_, cH, 0,
            bo + (size_t)l * cH, hf, nullptr, tmp, nullptr, cH, cH, 1.0f);
        ln_kernel<<<dim3((unsigned)TOK), 256, 0, stream>>>(
            tmp, ln1s + (size_t)l * cH, ln1b + (size_t)l * cH, hf, hb_);

        gemm_kernel<MODE_GELU><<<gF1, 256, 0, stream>>>(hb_, cH, 0, w1_, cH, 0,
            b1 + (size_t)l * cH4, nullptr, nullptr, nullptr, ffnb, cH, cH4, 1.0f);
        gemm_kernel<MODE_RESF32><<<gW, 256, 0, stream>>>(ffnb, cH4, 0, w2_, cH4, 0,
            b2 + (size_t)l * cH, hf, nullptr, tmp, nullptr, cH4, cH, 1.0f);
        ln_kernel<<<dim3((unsigned)TOK), 256, 0, stream>>>(
            tmp, ln2s + (size_t)l * cH, ln2b + (size_t)l * cH, hf, hb_);
    }

    head_kernel<<<dim3(cB), 64, 0, stream>>>(hf, annot, head_W, head_b, out);
    (void)in_sizes; (void)n_in; (void)out_size; (void)ws_size;
}
